// RNN_60069412602577
// MI455X (gfx1250) — matrix operand order repaired and checked by reading, not compiled, not run on hardware
//
#include <hip/hip_runtime.h>
#include <stdint.h>

#define N_B 64
#define T_S 512
#define D_I 1024
#define H_I 1024
#define NT  (N_B * T_S)   // 32768

typedef __attribute__((ext_vector_type(16))) __bf16          v16bf;
typedef __attribute__((ext_vector_type(16))) unsigned short  v16u;
typedef __attribute__((ext_vector_type(8)))  unsigned short  v8u;
typedef __attribute__((ext_vector_type(8)))  float           v8f;

// fp32 -> bf16, round-to-nearest-even
static __device__ __forceinline__ unsigned short f2bf(float f) {
    union { float f; unsigned u; } c; c.f = f;
    unsigned u = c.u;
    unsigned r = u + 0x7FFFu + ((u >> 16) & 1u);
    return (unsigned short)(r >> 16);
}

static __device__ __forceinline__ void store_bf16x4(unsigned short* p,
                                                    float a, float b, float c, float d) {
    unsigned lo = (unsigned)f2bf(a) | ((unsigned)f2bf(b) << 16);
    unsigned hi = (unsigned)f2bf(c) | ((unsigned)f2bf(d) << 16);
    unsigned long long v = (unsigned long long)lo | ((unsigned long long)hi << 32);
    *(unsigned long long*)p = v;
}

// tanh via hardware exp2/rcp: tanh(x) = (e^{2x}-1)/(e^{2x}+1).
// Clamp keeps e^{2x} finite (avoids inf/inf); |tanh(9)| is 1.0f in fp32 anyway.
static __device__ __forceinline__ float fast_tanh(float x) {
    float t = fminf(fmaxf(x, -9.0f), 9.0f);
    float e = __builtin_amdgcn_exp2f(t * 2.885390081777927f);   // e^(2t)
    return (e - 1.0f) * __builtin_amdgcn_rcpf(e + 1.0f);
}

// Load a 16x16x32 bf16 WMMA A-fragment: halves at byte offsets +0 and +32.
static __device__ __forceinline__ v16bf load_frag_split(const unsigned short* p) {
    v8u lo = *(const v8u*)(p);
    v8u hi = *(const v8u*)(p + 16);
    v16u w  = __builtin_shufflevector(lo, hi, 0,1,2,3,4,5,6,7,8,9,10,11,12,13,14,15);
    return __builtin_bit_cast(v16bf, w);
}

// Load a B-fragment whose 16 elements are fully contiguous.
static __device__ __forceinline__ v16bf load_frag_contig(const unsigned short* p) {
    v8u lo = *(const v8u*)(p);
    v8u hi = *(const v8u*)(p + 8);
    v16u w  = __builtin_shufflevector(lo, hi, 0,1,2,3,4,5,6,7,8,9,10,11,12,13,14,15);
    return __builtin_bit_cast(v16bf, w);
}

// ---------------------------------------------------------------------------
// Phase 0: transpose + convert weights: dst[n*1024 + k] = bf16(src[k*1024 + n])
// ---------------------------------------------------------------------------
__global__ __launch_bounds__(256)
void transpose_to_bf16(const float* __restrict__ src, unsigned short* __restrict__ dst) {
    __shared__ unsigned short tile[32][33];
    const int n0 = blockIdx.x * 32;
    const int k0 = blockIdx.y * 32;
    const int tid = threadIdx.x;
    const int r  = tid >> 3;        // 0..31
    const int c4 = (tid & 7) * 4;   // 0,4,...,28

    float4 v = *(const float4*)(src + (size_t)(k0 + r) * H_I + n0 + c4);
    tile[c4 + 0][r] = f2bf(v.x);
    tile[c4 + 1][r] = f2bf(v.y);
    tile[c4 + 2][r] = f2bf(v.z);
    tile[c4 + 3][r] = f2bf(v.w);
    __syncthreads();

    unsigned lo = (unsigned)tile[r][c4 + 0] | ((unsigned)tile[r][c4 + 1] << 16);
    unsigned hi = (unsigned)tile[r][c4 + 2] | ((unsigned)tile[r][c4 + 3] << 16);
    unsigned long long o = (unsigned long long)lo | ((unsigned long long)hi << 32);
    *(unsigned long long*)(dst + (size_t)(n0 + r) * D_I + k0 + c4) = o;
}

// ---------------------------------------------------------------------------
// Phase 1: out(NT x H) = x(NT x D) @ Wx(D x H) + b   [bf16 WMMA, fp32 accum]
// WG tile 128x128, 8 waves as 2(M)x4(N); wave tile 64x32 = 4x2 subtiles.
// ---------------------------------------------------------------------------
#define LDA 40   // padded A row stride in bf16 elements (80 B)

__global__ __launch_bounds__(256)
void xw_gemm(const float* __restrict__ x, const unsigned short* __restrict__ WxT,
             const float* __restrict__ bias, float* __restrict__ out) {
    __shared__ __align__(16) unsigned short sA[128 * LDA];

    const int tid  = threadIdx.x;
    const int lane = tid & 31;
    const int wave = tid >> 5;          // 0..7
    const int wm   = (wave >> 2) * 64;  // 0 or 64
    const int wn   = (wave & 3) * 32;   // 0..96
    const int lrow = lane & 15;
    const int hi16 = lane >> 4;
    const int kbA  = hi16 * 8;
    const int kbB  = hi16 * 16;
    const int m0   = blockIdx.x * 128;  // over NT
    const int n0   = blockIdx.y * 128;  // over H

    v8f acc[4][2] = {};

    const int sr = tid >> 3;            // staging row 0..31
    const int sc = (tid & 7) * 4;       // staging col 0..28

    for (int kk = 0; kk < D_I; kk += 32) {
        // stage 128x32 fp32 slice of x into LDS as bf16
        #pragma unroll
        for (int i = 0; i < 4; ++i) {
            const int row = sr + i * 32;
            float4 v = *(const float4*)(x + (size_t)(m0 + row) * D_I + kk + sc);
            store_bf16x4(&sA[row * LDA + sc], v.x, v.y, v.z, v.w);
        }
        __syncthreads();

        #pragma unroll
        for (int sn = 0; sn < 2; ++sn) {
            const int n = n0 + wn + sn * 16 + lrow;
            v16bf bf = load_frag_contig(WxT + (size_t)n * D_I + kk + kbB);
            #pragma unroll
            for (int sm = 0; sm < 4; ++sm) {
                v16bf af = load_frag_contig(&sA[(wm + sm * 16 + lrow) * LDA + kbB]);
                acc[sm][sn] = __builtin_amdgcn_wmma_f32_16x16x32_bf16(
                    false, af, false, bf, (short)0, acc[sm][sn], false, false);
            }
        }
        __syncthreads();
    }

    #pragma unroll
    for (int sn = 0; sn < 2; ++sn) {
        const int col = n0 + wn + sn * 16 + lrow;
        const float bv = bias[col];
        #pragma unroll
        for (int sm = 0; sm < 4; ++sm) {
            #pragma unroll
            for (int r = 0; r < 8; ++r) {
                const int row = m0 + wm + sm * 16 + r + hi16 * 8;
                out[(size_t)row * H_I + col] = acc[sm][sn][r] + bv;
            }
        }
    }
}

// ---------------------------------------------------------------------------
// Phase 2: recurrence. 4 WGs, one per 16 batch rows (independent -> no
// cross-WG sync). h slice (16x1024) lives in LDS as bf16; Wh stays in L2.
// ---------------------------------------------------------------------------
#define LDH 1032  // padded h row stride in bf16 elements (2064 B)

__global__ __launch_bounds__(1024)
void rnn_scan(const float* __restrict__ h0, const unsigned short* __restrict__ WhT,
              float* __restrict__ out) {
    __shared__ __align__(16) unsigned short sH[16 * LDH];

    const int tid  = threadIdx.x;
    const int lane = tid & 31;
    const int wave = tid >> 5;         // 0..31
    const int lrow = lane & 15;
    const int hi16 = lane >> 4;
    const int kbA  = hi16 * 8;
    const int kbB  = hi16 * 16;
    const int m0   = blockIdx.x * 16;  // batch rows owned by this WG
    const int wn   = wave * 32;        // H columns owned by this wave

    // Warm L2 with the full transposed Wh (read 512x during the scan).
    {
        const char* base = (const char*)WhT;
        const size_t total  = (size_t)H_I * H_I * sizeof(unsigned short); // 2 MB
        const size_t stride = (size_t)4096 * 128;                          // all scan threads
        for (size_t off = ((size_t)blockIdx.x * 1024 + tid) * 128; off < total; off += stride)
            __builtin_prefetch(base + off, 0, 3);
    }

    // init: h0 rows m0..m0+15 -> LDS bf16
    {
        const int r = tid >> 6;             // 0..15
        const int c = (tid & 63) * 16;      // 0..1008
        const float* p = h0 + (size_t)(m0 + r) * H_I + c;
        #pragma unroll
        for (int i = 0; i < 4; ++i) {
            float4 v = *(const float4*)(p + i * 4);
            store_bf16x4(&sH[r * LDH + c + i * 4], v.x, v.y, v.z, v.w);
        }
    }
    __syncthreads();

    #pragma unroll 1
    for (int t = 0; t < T_S; ++t) {
        v8f acc[2] = {};
        #pragma unroll 4
        for (int kk = 0; kk < H_I; kk += 32) {
            v16bf af = load_frag_contig(&sH[lrow * LDH + kk + kbB]);
            #pragma unroll
            for (int sn = 0; sn < 2; ++sn) {
                const int n = wn + sn * 16 + lrow;
                v16bf bf = load_frag_contig(WhT + (size_t)n * H_I + kk + kbB);
                acc[sn] = __builtin_amdgcn_wmma_f32_16x16x32_bf16(
                    false, af, false, bf, (short)0, acc[sn], false, false);
            }
        }
        __syncthreads();   // all waves done reading sH for this step

        #pragma unroll
        for (int sn = 0; sn < 2; ++sn) {
            const int col = wn + sn * 16 + lrow;
            #pragma unroll
            for (int r = 0; r < 8; ++r) {
                const int rl = r + hi16 * 8;                       // local batch row
                const size_t idx = ((size_t)(m0 + rl) * T_S + t) * H_I + col;
                const float v = fast_tanh(acc[sn][r] + out[idx]);  // xw_t read, h_t written
                out[idx] = v;
                sH[rl * LDH + col] = f2bf(v);                      // h for next step
            }
        }
        __syncthreads();   // sH fully updated before next step's reads
    }
}

// ---------------------------------------------------------------------------
extern "C" void kernel_launch(void* const* d_in, const int* in_sizes, int n_in,
                              void* d_out, int out_size, void* d_ws, size_t ws_size,
                              hipStream_t stream) {
    const float* x  = (const float*)d_in[0];
    const float* h0 = (const float*)d_in[1];
    const float* Wx = (const float*)d_in[2];
    const float* Wh = (const float*)d_in[3];
    const float* b  = (const float*)d_in[4];
    float* out = (float*)d_out;

    unsigned short* WxT = (unsigned short*)d_ws;             // 2 MB
    unsigned short* WhT = WxT + (size_t)D_I * H_I;           // 2 MB

    transpose_to_bf16<<<dim3(32, 32), 256, 0, stream>>>(Wx, WxT);
    transpose_to_bf16<<<dim3(32, 32), 256, 0, stream>>>(Wh, WhT);
    xw_gemm<<<dim3(NT / 128, H_I / 128), 256, 0, stream>>>(x, WxT, b, out);
    rnn_scan<<<dim3(N_B / 16), dim3(1024), 0, stream>>>(h0, WhT, out);
}
